// PointNet2_41540923687666
// MI455X (gfx1250) — matrix operand order repaired and checked by reading, not compiled, not run on hardware
//
#include <hip/hip_runtime.h>
#include <hip/hip_bf16.h>
#include <math.h>

// ---------------------------------------------------------------------------
// PointNet++ part-segmentation forward for MI455X (gfx1250, wave32, WMMA).
// All 1x1-conv MLP layers run as WMMA f16-in/f32-acc GEMMs (v_wmma_f32_16x16x32_f16).
// GEMM hot loop is branch-free: M is always a multiple of 64, W is packed with
// N padded to 16, so all loads are unconditional vector loads.
// ---------------------------------------------------------------------------

typedef _Float16 half_t;
typedef _Float16 v16h __attribute__((ext_vector_type(16)));
typedef float    v8f  __attribute__((ext_vector_type(8)));
typedef float    v4f  __attribute__((ext_vector_type(4)));

#define BATCH 8
#define NSAMP 32
#define BN_EPS 1e-5f

static inline int cdiv(int a, int b) { return (a + b - 1) / b; }
static inline int round32(int k) { return (k + 31) & ~31; }

// ---------------------------------------------------------------------------
// Weight prepack: f32 (N,K) row-major -> f16 (Np,Kp) row-major, zero-padded.
// ---------------------------------------------------------------------------
__global__ void pack_w_kernel(const float* __restrict__ W, int N, int K,
                              int Kp, int Np, half_t* __restrict__ Wp) {
  int tid = blockIdx.x * blockDim.x + threadIdx.x;
  if (tid >= Np * Kp) return;
  int k = tid % Kp, n = tid / Kp;
  Wp[tid] = (n < N && k < K) ? (half_t)W[(size_t)n * K + k] : (half_t)0.f;
}

// ---------------------------------------------------------------------------
// A-fragment loader: 16 f16 values for one lane of a 16x32 f16 A tile.
// lane<16 -> h0..7 = K(ka..ka+7), h8..15 = K(ka+16..ka+23)   (ka = k0 + hf*8)
// Unconditional vector loads (row always valid: M % 64 == 0).
// ---------------------------------------------------------------------------
__device__ __forceinline__ v16h load_a_frag(const float* __restrict__ xr, int ka) {
  v4f f0 = *(const v4f*)(xr + ka);
  v4f f1 = *(const v4f*)(xr + ka + 4);
  v4f f2 = *(const v4f*)(xr + ka + 16);
  v4f f3 = *(const v4f*)(xr + ka + 20);
  v16h a;
#pragma unroll
  for (int j = 0; j < 4; ++j) {
    a[j]      = (half_t)f0[j];
    a[4 + j]  = (half_t)f1[j];
    a[8 + j]  = (half_t)f2[j];
    a[12 + j] = (half_t)f3[j];
  }
  return a;
}

// ---------------------------------------------------------------------------
// WMMA GEMM: Y[M,N] = X[M,Kp](f32, lda=Kp) * Wp[Np,Kp]^T (f16) + bias.
// One wave per 64x16 output tile (4 stacked 16x16 WMMA tiles sharing B).
// Requires: M % 64 == 0, Kp % 32 == 0, Wp padded to Np=round16(N) rows.
// Layouts per CDNA5 ISA 7.12.2 (wave32):
//   A 16x32 f16 : lane<16 -> M=lane, h0..7=K(k0..+7), h8..15=K(k0+16..+23)
//                 lane>=16-> M=lane-16, K bases +8 / +24
//   B 32x16 f16 : lane<16 -> N=lane, h0..7=K(k0..+7), h8..15=K(k0+16..+23); lane>=16 -> +8
//   C/D 16x16 f32: VGPR r -> M = r + 8*(lane>=16), N = lane&15
// ---------------------------------------------------------------------------
__global__ __launch_bounds__(32)
void wmma_gemm_bias_kernel(const float* __restrict__ X, int M, int Kp,
                           const half_t* __restrict__ Wp, const float* __restrict__ bias,
                           int N, float* __restrict__ Y, int ldy) {
  const int lane = threadIdx.x & 31;
  const int hf   = lane >> 4;      // half-wave: 0 or 1
  const int l    = lane & 15;
  const int m0   = blockIdx.x * 64;
  const int n0   = blockIdx.y * 16;

  const float* xr0 = X + (size_t)(m0 +  0 + l) * Kp;
  const float* xr1 = X + (size_t)(m0 + 16 + l) * Kp;
  const float* xr2 = X + (size_t)(m0 + 32 + l) * Kp;
  const float* xr3 = X + (size_t)(m0 + 48 + l) * Kp;
  const half_t* wrow = Wp + (size_t)(n0 + l) * Kp;   // always valid (Np padded)

  v8f acc0 = {}, acc1 = {}, acc2 = {}, acc3 = {};
  for (int k0 = 0; k0 < Kp; k0 += 32) {
    const int ka = k0 + hf * 8;
    const int kb = k0 + hf * 16;
    typedef _Float16 v8h_ __attribute__((ext_vector_type(8)));
    v8h_ blo = *(const v8h_*)(wrow + ka);
    v8h_ bhi = *(const v8h_*)(wrow + ka + 16);
    v16h b = __builtin_shufflevector(blo, bhi, 0,1,2,3,4,5,6,7,8,9,10,11,12,13,14,15);
    v16h a0 = load_a_frag(xr0, ka);
    v16h a1 = load_a_frag(xr1, ka);
    v16h a2 = load_a_frag(xr2, ka);
    v16h a3 = load_a_frag(xr3, ka);
    acc0 = __builtin_amdgcn_wmma_f32_16x16x32_f16(false, a0, false, b, (short)0, acc0, false, false);
    acc1 = __builtin_amdgcn_wmma_f32_16x16x32_f16(false, a1, false, b, (short)0, acc1, false, false);
    acc2 = __builtin_amdgcn_wmma_f32_16x16x32_f16(false, a2, false, b, (short)0, acc2, false, false);
    acc3 = __builtin_amdgcn_wmma_f32_16x16x32_f16(false, a3, false, b, (short)0, acc3, false, false);
  }

  const int col = n0 + l;
  if (col < N) {                       // only head2 (N=5) ever takes the miss
    const float bv = bias ? bias[col] : 0.f;
    float* y0 = Y + (size_t)(m0 +  0 + hf * 8) * ldy + col;
    float* y1 = Y + (size_t)(m0 + 16 + hf * 8) * ldy + col;
    float* y2 = Y + (size_t)(m0 + 32 + hf * 8) * ldy + col;
    float* y3 = Y + (size_t)(m0 + 48 + hf * 8) * ldy + col;
#pragma unroll
    for (int r = 0; r < 8; ++r) {
      y0[(size_t)r * ldy] = acc0[r] + bv;
      y1[(size_t)r * ldy] = acc1[r] + bv;
      y2[(size_t)r * ldy] = acc2[r] + bv;
      y3[(size_t)r * ldy] = acc3[r] + bv;
    }
  }
}

// ---------------------------------------------------------------------------
// BatchNorm (training mode: batch stats over rows) -- two-pass.
// ---------------------------------------------------------------------------
__global__ void bn_stats_kernel(const float* __restrict__ Y, int M, int C,
                                float* __restrict__ mean, float* __restrict__ var) {
  __shared__ float ss[256], sq[256];
  const int ch = blockIdx.x, t = threadIdx.x;
  float s = 0.f, q = 0.f;
  for (int i = t; i < M; i += 256) {
    float v = Y[(size_t)i * C + ch];
    s += v; q += v * v;
  }
  ss[t] = s; sq[t] = q;
  __syncthreads();
  for (int st = 128; st > 0; st >>= 1) {
    if (t < st) { ss[t] += ss[t + st]; sq[t] += sq[t + st]; }
    __syncthreads();
  }
  if (t == 0) {
    float mu = ss[0] / (float)M;
    mean[ch] = mu;
    var[ch]  = fmaxf(sq[0] / (float)M - mu * mu, 0.f);
  }
}

__global__ void bn_relu_kernel(float* __restrict__ Y, int M, int C,
                               const float* __restrict__ mean, const float* __restrict__ var,
                               const float* __restrict__ g, const float* __restrict__ beta) {
  size_t tid = (size_t)blockIdx.x * blockDim.x + threadIdx.x;
  size_t total = (size_t)M * C;
  if (tid >= total) return;
  int ch = (int)(tid % C);
  float v = Y[tid];
  float r = g[ch] * (v - mean[ch]) / sqrtf(var[ch] + BN_EPS) + beta[ch];
  Y[tid] = fmaxf(r, 0.f);
}

// ---------------------------------------------------------------------------
// Farthest point sampling; matches reference (start index 0, first-max ties).
// One workgroup per batch; running min-distance array lives in LDS.
// ---------------------------------------------------------------------------
__global__ void fps_kernel(const float* __restrict__ xyz, int Nsrc, int npoint,
                           float* __restrict__ out_xyz) {
  __shared__ float sdist[4096];
  __shared__ float rv[256];
  __shared__ int   ri[256];
  __shared__ float cx, cy, cz;
  __shared__ int   sfar;
  const int b = blockIdx.x, t = threadIdx.x;
  const float* p = xyz + (size_t)b * Nsrc * 3;
  for (int i = t; i < Nsrc; i += 256) sdist[i] = 1e10f;
  if (t == 0) sfar = 0;
  __syncthreads();
  for (int it = 0; it < npoint; ++it) {
    if (t == 0) {
      int f = sfar;
      cx = p[f * 3 + 0]; cy = p[f * 3 + 1]; cz = p[f * 3 + 2];
      float* o = out_xyz + ((size_t)b * npoint + it) * 3;
      o[0] = cx; o[1] = cy; o[2] = cz;
    }
    __syncthreads();
    float best = -1.f; int bi = 0;
    for (int i = t; i < Nsrc; i += 256) {
      float dx = p[i * 3] - cx, dy = p[i * 3 + 1] - cy, dz = p[i * 3 + 2] - cz;
      float d  = dx * dx + dy * dy + dz * dz;
      float nd = fminf(sdist[i], d);
      sdist[i] = nd;
      if (nd > best) { best = nd; bi = i; }  // strided ascending -> first max kept
    }
    rv[t] = best; ri[t] = bi;
    __syncthreads();
    for (int s = 128; s > 0; s >>= 1) {
      if (t < s) {
        if (rv[t + s] > rv[t] || (rv[t + s] == rv[t] && ri[t + s] < ri[t])) {
          rv[t] = rv[t + s]; ri[t] = ri[t + s];
        }
      }
      __syncthreads();
    }
    if (t == 0) sfar = ri[0];
    __syncthreads();
  }
}

// ---------------------------------------------------------------------------
// kNN (k=32): one thread per query center; unrolled insertion sort.
// Strict '<' insert reproduces top-k tie-breaking by lower index.
// ---------------------------------------------------------------------------
__global__ void knn32_kernel(const float* __restrict__ q, int nq,
                             const float* __restrict__ src, int Nsrc,
                             int* __restrict__ idx, int total) {
  int tid = blockIdx.x * blockDim.x + threadIdx.x;
  if (tid >= total) return;
  int b = tid / nq;
  const float* Q = q + (size_t)tid * 3;
  const float qx = Q[0], qy = Q[1], qz = Q[2];
  const float* S = src + (size_t)b * Nsrc * 3;
  float bd[32]; int bix[32];
#pragma unroll
  for (int j = 0; j < 32; ++j) { bd[j] = 3.4e38f; bix[j] = 0; }
  for (int i = 0; i < Nsrc; ++i) {
    float dx = S[i * 3] - qx, dy = S[i * 3 + 1] - qy, dz = S[i * 3 + 2] - qz;
    float d = dx * dx + dy * dy + dz * dz;
    if (d < bd[31]) {
      int pos = 31;
#pragma unroll
      for (int j = 30; j >= 0; --j) {
        if (d < bd[j]) { bd[j + 1] = bd[j]; bix[j + 1] = bix[j]; pos = j; }
      }
      bd[pos] = d; bix[pos] = i;
    }
  }
  int* o = idx + (size_t)tid * 32;
#pragma unroll
  for (int j = 0; j < 32; ++j) o[j] = bix[j];
}

// ---------------------------------------------------------------------------
// SA grouping: build GEMM rows [centered xyz(3) | gathered feats(C) | 0-pad].
// ---------------------------------------------------------------------------
__global__ void sa_gather_kernel(const float* __restrict__ src_xyz,
                                 const float* __restrict__ src_pts, int Nsrc, int C,
                                 const float* __restrict__ new_xyz,
                                 const int* __restrict__ idx, int np, int Kp,
                                 float* __restrict__ X, int total) {
  int tid = blockIdx.x * blockDim.x + threadIdx.x;
  if (tid >= total) return;
  int c = (tid / NSAMP) % np;
  int b = tid / (NSAMP * np);
  int i = idx[tid];
  const float* sx = src_xyz + ((size_t)b * Nsrc + i) * 3;
  const float* nx = new_xyz + ((size_t)b * np + c) * 3;
  float* row = X + (size_t)tid * Kp;
  row[0] = sx[0] - nx[0]; row[1] = sx[1] - nx[1]; row[2] = sx[2] - nx[2];
  const float* sp = src_pts + ((size_t)b * Nsrc + i) * C;
  for (int j = 0; j < C; ++j) row[3 + j] = sp[j];
  for (int j = 3 + C; j < Kp; ++j) row[j] = 0.f;
}

// ---------------------------------------------------------------------------
// Max-pool over the nsample axis.
// ---------------------------------------------------------------------------
__global__ void maxpool_kernel(const float* __restrict__ Y, int npts, int C,
                               float* __restrict__ out) {
  int tid = blockIdx.x * blockDim.x + threadIdx.x;
  if (tid >= npts * C) return;
  int ch = tid % C, p = tid / C;
  const float* base = Y + ((size_t)p * NSAMP) * C + ch;
  float m = -3.4e38f;
  for (int s = 0; s < NSAMP; ++s) m = fmaxf(m, base[(size_t)s * C]);
  out[(size_t)p * C + ch] = m;
}

// ---------------------------------------------------------------------------
// FP: 3-NN inverse-distance interpolation + skip concat into GEMM rows.
// Row = [pts1(C1) | interp(C2) | 0-pad to Kp]
// ---------------------------------------------------------------------------
__global__ void fp_gather_kernel(const float* __restrict__ xyz1, int N1,
                                 const float* __restrict__ xyz2, int N2,
                                 const float* __restrict__ pts1, int C1,
                                 const float* __restrict__ pts2, int C2,
                                 int Kp, float* __restrict__ X, int total) {
  int tid = blockIdx.x * blockDim.x + threadIdx.x;
  if (tid >= total) return;
  int b = tid / N1;
  const float* Q = xyz1 + (size_t)tid * 3;
  const float qx = Q[0], qy = Q[1], qz = Q[2];
  const float* S = xyz2 + (size_t)b * N2 * 3;
  float d0 = 3.4e38f, d1 = 3.4e38f, d2 = 3.4e38f;
  int   i0 = 0, i1 = 0, i2 = 0;
  for (int i = 0; i < N2; ++i) {
    float dx = S[i * 3] - qx, dy = S[i * 3 + 1] - qy, dz = S[i * 3 + 2] - qz;
    float d = dx * dx + dy * dy + dz * dz;
    if (d < d0)      { d2 = d1; i2 = i1; d1 = d0; i1 = i0; d0 = d; i0 = i; }
    else if (d < d1) { d2 = d1; i2 = i1; d1 = d;  i1 = i; }
    else if (d < d2) { d2 = d;  i2 = i; }
  }
  float w0 = 1.f / (sqrtf(fmaxf(d0, 0.f)) + 1e-8f);
  float w1 = 1.f / (sqrtf(fmaxf(d1, 0.f)) + 1e-8f);
  float w2 = 1.f / (sqrtf(fmaxf(d2, 0.f)) + 1e-8f);
  float ws = w0 + w1 + w2;
  w0 /= ws; w1 /= ws; w2 /= ws;

  float* row = X + (size_t)tid * Kp;
  const float* p1 = pts1 + (size_t)tid * C1;
  for (int j = 0; j < C1; ++j) row[j] = p1[j];
  const float* P = pts2 + (size_t)b * N2 * C2;
  for (int j = 0; j < C2; ++j)
    row[C1 + j] = w0 * P[(size_t)i0 * C2 + j] + w1 * P[(size_t)i1 * C2 + j]
                + w2 * P[(size_t)i2 * C2 + j];
  for (int j = C1 + C2; j < Kp; ++j) row[j] = 0.f;
}

// ---------------------------------------------------------------------------
// Host-side orchestration helpers
// ---------------------------------------------------------------------------
static void linear_layer(const float* X, int M, int K, int Kp,
                         const float* W, const float* bias, int N,
                         float* Y, int ldy, half_t* wpack, hipStream_t st) {
  int Np = (N + 15) & ~15;
  int tot = Np * Kp;
  pack_w_kernel<<<cdiv(tot, 256), 256, 0, st>>>(W, N, K, Kp, Np, wpack);
  dim3 grid(M / 64, Np / 16);        // M always a multiple of 64 in this net
  wmma_gemm_bias_kernel<<<grid, 32, 0, st>>>(X, M, Kp, wpack, bias, N, Y, ldy);
}

static void bn_relu(float* Y, int M, int C, const float* g, const float* beta,
                    float* mean, float* var, hipStream_t st) {
  bn_stats_kernel<<<C, 256, 0, st>>>(Y, M, C, mean, var);
  size_t tot = (size_t)M * C;
  bn_relu_kernel<<<cdiv((int)tot, 256), 256, 0, st>>>(Y, M, C, mean, var, g, beta);
}

// MLP chain (linear + BN(batch stats) + ReLU per layer) with ping-pong buffers.
// Input rows must be in bufA with lda = round32(K0). If fout != nullptr the
// final layer is written (and normalized) directly into fout.
static const float* mlp_chain(float* bufA, float* bufB, int M, int K0,
                              void* const* d_in, int base, int nlayers, const int* couts,
                              half_t* wpack, float* mean, float* var,
                              float* fout, hipStream_t st) {
  float* cur = bufA;
  float* nxt = bufB;
  int K = K0, Kp = round32(K0);
  for (int li = 0; li < nlayers; ++li) {
    const float* W  = (const float*)d_in[base + li * 4 + 0];
    const float* b  = (const float*)d_in[base + li * 4 + 1];
    const float* g  = (const float*)d_in[base + li * 4 + 2];
    const float* bt = (const float*)d_in[base + li * 4 + 3];
    int N = couts[li];
    float* dst = (li == nlayers - 1 && fout) ? fout : nxt;
    linear_layer(cur, M, K, Kp, W, b, N, dst, N, wpack, st);
    bn_relu(dst, M, N, g, bt, mean, var, st);
    float* scratch = (dst == nxt) ? cur : nxt;
    cur = dst; nxt = scratch;
    K = N; Kp = N;  // all intermediate widths are multiples of 32
  }
  return cur;
}

static void sa_level(const float* src_xyz, int Nsrc, const float* src_pts, int C,
                     int np, void* const* d_in, int base, const int* couts, int nl,
                     float* new_xyz, int* idx, float* out_pts,
                     float* bufA, float* bufB, half_t* wpack,
                     float* mean, float* var, hipStream_t st) {
  fps_kernel<<<BATCH, 256, 0, st>>>(src_xyz, Nsrc, np, new_xyz);
  int totq = BATCH * np;
  knn32_kernel<<<cdiv(totq, 128), 128, 0, st>>>(new_xyz, np, src_xyz, Nsrc, idx, totq);
  int K0 = 3 + C, Kp0 = round32(K0);
  int rows = BATCH * np * NSAMP;
  sa_gather_kernel<<<cdiv(rows, 256), 256, 0, st>>>(src_xyz, src_pts, Nsrc, C,
                                                    new_xyz, idx, np, Kp0, bufA, rows);
  const float* last = mlp_chain(bufA, bufB, rows, K0, d_in, base, nl, couts,
                                wpack, mean, var, nullptr, st);
  int Cl = couts[nl - 1];
  int tot = BATCH * np * Cl;
  maxpool_kernel<<<cdiv(tot, 256), 256, 0, st>>>(last, BATCH * np, Cl, out_pts);
}

static void fp_level(const float* xyz1, int N1, const float* xyz2, int N2,
                     const float* pts1, int C1, const float* pts2, int C2,
                     void* const* d_in, int base, const int* couts, int nl,
                     float* out, float* bufA, float* bufB, half_t* wpack,
                     float* mean, float* var, hipStream_t st) {
  int K0 = C1 + C2, Kp0 = round32(K0);
  int tot = BATCH * N1;
  fp_gather_kernel<<<cdiv(tot, 128), 128, 0, st>>>(xyz1, N1, xyz2, N2, pts1, C1,
                                                   pts2, C2, Kp0, bufA, tot);
  mlp_chain(bufA, bufB, tot, K0, d_in, base, nl, couts,
            wpack, mean, var, out, st);
}

// ---------------------------------------------------------------------------
// kernel_launch: full forward pass.
// Input order (recursive dict insertion order): coords, features, then params:
//   sa1..sa4 (3 layers x {W,b,g,beta}), fp4,fp3,fp2 (2 layers), fp1 (3 layers),
//   head1 {W,b,g,beta}, head2 {W,b,g,beta}.
// ---------------------------------------------------------------------------
extern "C" void kernel_launch(void* const* d_in, const int* in_sizes, int n_in,
                              void* d_out, int out_size, void* d_ws, size_t ws_size,
                              hipStream_t stream) {
  (void)in_sizes; (void)n_in; (void)out_size; (void)ws_size;

  const float* coords   = (const float*)d_in[0];  // (8,4096,3)
  const float* features = (const float*)d_in[1];  // (8,4096,6)

  // ---- workspace carve (floats, 256B-aligned chunks) ----
  float* ws = (float*)d_ws;
  size_t off = 0;
  auto alloc = [&](size_t n) -> float* {
    float* p = ws + off;
    off += (n + 63) & ~(size_t)63;
    return p;
  };
  float* bufA   = alloc((size_t)262144 * 64);  // 64 MB ping
  float* bufB   = alloc((size_t)262144 * 64);  // 64 MB pong
  half_t* wpack = (half_t*)alloc(98304);       // max 256x768 f16 packed weights
  float* mean   = alloc(512);
  float* var    = alloc(512);
  float* l1x = alloc((size_t)BATCH * 1024 * 3);
  int*   l1i = (int*)alloc((size_t)BATCH * 1024 * 32);
  float* l1p = alloc((size_t)BATCH * 1024 * 64);
  float* l2x = alloc((size_t)BATCH * 256 * 3);
  int*   l2i = (int*)alloc((size_t)BATCH * 256 * 32);
  float* l2p = alloc((size_t)BATCH * 256 * 128);
  float* l3x = alloc((size_t)BATCH * 64 * 3);
  int*   l3i = (int*)alloc((size_t)BATCH * 64 * 32);
  float* l3p = alloc((size_t)BATCH * 64 * 256);
  float* l4x = alloc((size_t)BATCH * 16 * 3);
  int*   l4i = (int*)alloc((size_t)BATCH * 16 * 32);
  float* l4p = alloc((size_t)BATCH * 16 * 512);
  float* fp4o = alloc((size_t)BATCH * 64 * 256);
  float* fp3o = alloc((size_t)BATCH * 256 * 256);
  float* fp2o = alloc((size_t)BATCH * 1024 * 128);
  float* fp1o = alloc((size_t)BATCH * 4096 * 128);

  static const int sa1_c[3] = {32, 32, 64};
  static const int sa2_c[3] = {64, 64, 128};
  static const int sa3_c[3] = {128, 128, 256};
  static const int sa4_c[3] = {256, 256, 512};
  static const int fp4_c[2] = {256, 256};
  static const int fp3_c[2] = {256, 256};
  static const int fp2_c[2] = {256, 128};
  static const int fp1_c[3] = {128, 128, 128};

  // ---- set abstraction ----
  sa_level(coords, 4096, features,   6, 1024, d_in,  2, sa1_c, 3, l1x, l1i, l1p,
           bufA, bufB, wpack, mean, var, stream);
  sa_level(l1x,    1024, l1p,       64,  256, d_in, 14, sa2_c, 3, l2x, l2i, l2p,
           bufA, bufB, wpack, mean, var, stream);
  sa_level(l2x,     256, l2p,      128,   64, d_in, 26, sa3_c, 3, l3x, l3i, l3p,
           bufA, bufB, wpack, mean, var, stream);
  sa_level(l3x,      64, l3p,      256,   16, d_in, 38, sa4_c, 3, l4x, l4i, l4p,
           bufA, bufB, wpack, mean, var, stream);

  // ---- feature propagation ----
  fp_level(l3x,   64, l4x,   16, l3p,      256, l4p,  512, d_in, 50, fp4_c, 2, fp4o,
           bufA, bufB, wpack, mean, var, stream);
  fp_level(l2x,  256, l3x,   64, l2p,      128, fp4o, 256, d_in, 58, fp3_c, 2, fp3o,
           bufA, bufB, wpack, mean, var, stream);
  fp_level(l1x, 1024, l2x,  256, l1p,       64, fp3o, 256, d_in, 66, fp2_c, 2, fp2o,
           bufA, bufB, wpack, mean, var, stream);
  fp_level(coords, 4096, l1x, 1024, features, 6, fp2o, 128, d_in, 74, fp1_c, 3, fp1o,
           bufA, bufB, wpack, mean, var, stream);

  // ---- head ----
  const int Mh = BATCH * 4096;
  const float* h1W = (const float*)d_in[86];
  const float* h1b = (const float*)d_in[87];
  const float* h1g = (const float*)d_in[88];
  const float* h1t = (const float*)d_in[89];
  const float* h2W = (const float*)d_in[90];
  const float* h2b = (const float*)d_in[91];

  linear_layer(fp1o, Mh, 128, 128, h1W, h1b, 128, bufA, 128, wpack, stream);
  bn_relu(bufA, Mh, 128, h1g, h1t, mean, var, stream);
  linear_layer(bufA, Mh, 128, 128, h2W, h2b, 5, (float*)d_out, 5, wpack, stream);
}
